// NeuralHSMM_83872121356744
// MI455X (gfx1250) — matrix operand order repaired and checked by reading, not compiled, not run on hardware
//
#include <hip/hip_runtime.h>
#include <hip/hip_bf16.h>
#include <math.h>

typedef __attribute__((ext_vector_type(2)))  float    v2f;
typedef __attribute__((ext_vector_type(8)))  float    v8f;
typedef __attribute__((ext_vector_type(16))) _Float16 v16h;
typedef __attribute__((ext_vector_type(4)))  int      v4i;

#define BB    1024
#define LL    2048
#define CC    12
#define HH    128
#define NG    48      // 4*C
#define KC    24      // 2*C
#define NMODE 4
#define NT    7

// ---- gfx1250 async global->LDS copy (double-buffered x prefetch) -----------
#ifdef __has_builtin
# if __has_builtin(__builtin_amdgcn_global_load_async_to_lds_b128)
#  define HAVE_ASYNC_CP 1
# endif
#endif
#ifndef HAVE_ASYNC_CP
# define HAVE_ASYNC_CP 0
#endif

#if HAVE_ASYNC_CP
typedef __attribute__((address_space(1))) v4i* as1_v4i_p;   // global v4i*
typedef __attribute__((address_space(3))) v4i* as3_v4i_p;   // LDS v4i*
__device__ __forceinline__ void async_cp16(const float* g, float* l) {
  __builtin_amdgcn_global_load_async_to_lds_b128((as1_v4i_p)g, (as3_v4i_p)l, 0, 0);
}
__device__ __forceinline__ void wait_async_le2() {
# if __has_builtin(__builtin_amdgcn_s_wait_asynccnt)
  __builtin_amdgcn_s_wait_asynccnt(2);
# else
  asm volatile("s_wait_asynccnt 2" ::: "memory");
# endif
}
#endif

// ---- branch-free fast activations on the TRANS32 unit ----------------------
__device__ __forceinline__ float fexp_(float x)  { return __builtin_amdgcn_exp2f(x * 1.44269504088896340736f); }
__device__ __forceinline__ float frcp_(float x)  { return __builtin_amdgcn_rcpf(x); }
__device__ __forceinline__ float fsig_(float x)  { return frcp_(1.f + __builtin_amdgcn_exp2f(-1.44269504088896340736f * x)); }
__device__ __forceinline__ float ftanh_(float x) {
  const float t = __builtin_amdgcn_exp2f(-2.88539008177792681472f * fabsf(x)); // exp(-2|x|)
  const float r = (1.f - t) * frcp_(1.f + t);
  return copysignf(r, x);
}

// ---------------------------------------------------------------------------
// Kernel 1: local tanh-linear + 2-layer LSTM scan, fused, 16 batch rows/wave.
// gates(16x48) = [xl|h](16x24) @ Wcat^T(24x48) via 18x V_WMMA_F32_16X16X4_F32.
// Weights live in VGPRs for the whole scan; state lives in LDS; next step's
// x rows stream in via GLOBAL_LOAD_ASYNC_TO_LDS_B128 (ASYNCcnt pipelined).
// ---------------------------------------------------------------------------
__global__ __launch_bounds__(32) void k_lstm_scan(
    const float* __restrict__ x, const int* __restrict__ lengths,
    const float* __restrict__ localW, const float* __restrict__ localb,
    const float* __restrict__ Wih0, const float* __restrict__ Whh0,
    const float* __restrict__ bih0, const float* __restrict__ bhh0,
    const float* __restrict__ Wih1, const float* __restrict__ Whh1,
    const float* __restrict__ bih1, const float* __restrict__ bhh1,
    float* __restrict__ enc)
{
  const int lane = threadIdx.x & 31;
  const int m    = lane & 15;
  const int hi   = lane >> 4;
  const int row0 = blockIdx.x * 16;

  __shared__ float ls_x [2][16][CC];  // double-buffered x staging
  __shared__ float ls_v0[16][KC];     // [xl | h0_prev]
  __shared__ float ls_v1[16][KC];     // [h0  | h1_prev]
  __shared__ float ls_c0[16][CC];
  __shared__ float ls_c1[16][CC];
  __shared__ float ls_g [16][NG];
  __shared__ int   ls_len[16];

  for (int i = lane; i < 16*KC; i += 32) { (&ls_v0[0][0])[i] = 0.f; (&ls_v1[0][0])[i] = 0.f; }
  for (int i = lane; i < 16*CC; i += 32) { (&ls_c0[0][0])[i] = 0.f; (&ls_c1[0][0])[i] = 0.f; }
  if (lane < 16) ls_len[lane] = lengths[row0 + lane];
  __syncthreads();

  // ---- pack B-operands (W^T tiles) into VGPRs, resident for whole scan ----
  // f32 4x16 B tile: elem0 -> K = 4k+2*hi, elem1 -> K = 4k+2*hi+1, N = m.
  v2f BL[3];
  #pragma unroll
  for (int k = 0; k < 3; ++k) {
    const int kb = 4*k + 2*hi;
    float e0 = 0.f, e1 = 0.f;
    if (m < CC) { e0 = localW[m*CC + kb]; e1 = localW[m*CC + kb + 1]; }
    BL[k][0] = e0; BL[k][1] = e1;
  }
  v2f B0[6][3], B1[6][3];
  #pragma unroll
  for (int k = 0; k < 6; ++k) {
    #pragma unroll
    for (int nt = 0; nt < 3; ++nt) {
      const int n  = nt*16 + m;           // always < 48
      const int kb = 4*k + 2*hi;
      B0[k][nt][0] = (kb   < CC) ? Wih0[n*CC + kb]     : Whh0[n*CC + (kb   - CC)];
      B0[k][nt][1] = (kb+1 < CC) ? Wih0[n*CC + kb + 1] : Whh0[n*CC + (kb+1 - CC)];
      B1[k][nt][0] = (kb   < CC) ? Wih1[n*CC + kb]     : Whh1[n*CC + (kb   - CC)];
      B1[k][nt][1] = (kb+1 < CC) ? Wih1[n*CC + kb + 1] : Whh1[n*CC + (kb+1 - CC)];
    }
  }
  const float blv = (m < CC) ? localb[m] : 0.f;
  float bias0[3], bias1[3];
  #pragma unroll
  for (int nt = 0; nt < 3; ++nt) {
    const int n = nt*16 + m;
    bias0[nt] = bih0[n] + bhh0[n];
    bias1[nt] = bih1[n] + bhh1[n];
  }

#if HAVE_ASYNC_CP
  // 48 chunks of 16B cover the 16x12 f32 tile; lane -> chunk, lanes<16 -> chunk+32
  const int rA = lane / 3,              pA = lane % 3;
  const int rB = (lane < 16) ? (32 + lane) / 3 : 0;
  const int pB = (lane < 16) ? (32 + lane) % 3 : 0;
  const float* gA = x + (size_t)(row0 + rA)*LL*CC + pA*4;
  const float* gB = x + (size_t)(row0 + rB)*LL*CC + pB*4;
  // prologue: t = 0 into buffer 0
  async_cp16(gA, &ls_x[0][rA][pA*4]);
  if (lane < 16) async_cp16(gB, &ls_x[0][rB][pB*4]);
#endif

  for (int t = 0; t < LL; ++t) {
    const int buf = t & 1;

#if HAVE_ASYNC_CP
    // 1) issue async fetch of next step (clamped), then retire current buffer
    {
      const int tn = (t + 1 < LL) ? (t + 1) : (LL - 1);
      const int bn = (t + 1) & 1;
      async_cp16(gA + (size_t)tn*CC, &ls_x[bn][rA][pA*4]);
      if (lane < 16) async_cp16(gB + (size_t)tn*CC, &ls_x[bn][rB][pB*4]);
      wait_async_le2();   // oldest pair (buffer for step t) is complete
    }
#else
    // 1) synchronous fetch of this timestep's x rows (48B/row, 16B-aligned)
    if (lane < 16) {
      const float4* px = reinterpret_cast<const float4*>(x + ((size_t)(row0 + lane)*LL + t)*CC);
      float4 a = px[0], b = px[1], c = px[2];
      float4* dst = reinterpret_cast<float4*>(&ls_x[buf][lane][0]);
      dst[0] = a; dst[1] = b; dst[2] = c;
    }
#endif
    __syncthreads();

    // 2) xl = tanh(x @ localW^T + b)   (3x WMMA f32 16x16x4)
    {
      v8f acc;
      #pragma unroll
      for (int r = 0; r < 8; ++r) acc[r] = blv;
      #pragma unroll
      for (int k = 0; k < 3; ++k) {
        const int kb = 4*k + 2*hi;
        v2f a; a[0] = ls_x[buf][m][kb]; a[1] = ls_x[buf][m][kb+1];
        acc = __builtin_amdgcn_wmma_f32_16x16x4_f32(false, a, false, BL[k], (short)0, acc, false, false);
      }
      #pragma unroll
      for (int r = 0; r < 8; ++r) {
        const int M = r + 8*hi;
        if (m < CC) ls_v0[M][m] = ftanh_(acc[r]);
      }
    }
    __syncthreads();

    // 3) layer-0 gates (18x WMMA)
    {
      v2f a[6];
      #pragma unroll
      for (int k = 0; k < 6; ++k) { const int kb = 4*k + 2*hi; a[k][0] = ls_v0[m][kb]; a[k][1] = ls_v0[m][kb+1]; }
      #pragma unroll
      for (int nt = 0; nt < 3; ++nt) {
        v8f g;
        #pragma unroll
        for (int r = 0; r < 8; ++r) g[r] = bias0[nt];
        #pragma unroll
        for (int k = 0; k < 6; ++k)
          g = __builtin_amdgcn_wmma_f32_16x16x4_f32(false, a[k], false, B0[k][nt], (short)0, g, false, false);
        #pragma unroll
        for (int r = 0; r < 8; ++r) ls_g[r + 8*hi][nt*16 + m] = g[r];
      }
    }
    __syncthreads();

    // 4) layer-0 cell update (6 cells / lane, branch-free activations)
    #pragma unroll
    for (int j = 0; j < 6; ++j) {
      const int idx = lane*6 + j, rr = idx / CC, ch = idx % CC;
      const float ig = fsig_ (ls_g[rr][ch]);
      const float fg = fsig_ (ls_g[rr][ch + CC]);
      const float gg = ftanh_(ls_g[rr][ch + 2*CC]);
      const float og = fsig_ (ls_g[rr][ch + 3*CC]);
      const float c  = fg*ls_c0[rr][ch] + ig*gg;
      const float h  = og*ftanh_(c);
      ls_c0[rr][ch]      = c;
      ls_v0[rr][CC + ch] = h;   // h0 for next step's layer 0
      ls_v1[rr][ch]      = h;   // layer-1 input now
    }
    __syncthreads();

    // 5) layer-1 gates (18x WMMA)
    {
      v2f a[6];
      #pragma unroll
      for (int k = 0; k < 6; ++k) { const int kb = 4*k + 2*hi; a[k][0] = ls_v1[m][kb]; a[k][1] = ls_v1[m][kb+1]; }
      #pragma unroll
      for (int nt = 0; nt < 3; ++nt) {
        v8f g;
        #pragma unroll
        for (int r = 0; r < 8; ++r) g[r] = bias1[nt];
        #pragma unroll
        for (int k = 0; k < 6; ++k)
          g = __builtin_amdgcn_wmma_f32_16x16x4_f32(false, a[k], false, B1[k][nt], (short)0, g, false, false);
        #pragma unroll
        for (int r = 0; r < 8; ++r) ls_g[r + 8*hi][nt*16 + m] = g[r];
      }
    }
    __syncthreads();

    // 6) layer-1 cell update + masked enc store
    #pragma unroll
    for (int j = 0; j < 6; ++j) {
      const int idx = lane*6 + j, rr = idx / CC, ch = idx % CC;
      const float ig = fsig_ (ls_g[rr][ch]);
      const float fg = fsig_ (ls_g[rr][ch + CC]);
      const float gg = ftanh_(ls_g[rr][ch + 2*CC]);
      const float og = fsig_ (ls_g[rr][ch + 3*CC]);
      const float c  = fg*ls_c1[rr][ch] + ig*gg;
      const float h  = og*ftanh_(c);
      ls_c1[rr][ch]      = c;
      ls_v1[rr][CC + ch] = h;
      const float e = (t < ls_len[rr]) ? h : 0.f;
      enc[((size_t)(row0 + rr)*LL + t)*CC + ch] = e;
    }
    __syncthreads();
  }
}

// ---------------------------------------------------------------------------
// Kernel 2: attention scores, fused feat->128->tanh->1 MLP.
// K padded 13->32, one v_wmma_f32_16x16x32_f16 per 16-col tile of W1^T.
// Hidden layer never touches memory; W2-dot via shfl_xor butterfly.
// ---------------------------------------------------------------------------
__global__ __launch_bounds__(32) void k_attn_score(
    const float* __restrict__ enc, const int* __restrict__ lengths,
    const float* __restrict__ W1, const float* __restrict__ b1,
    const float* __restrict__ W2, const float* __restrict__ b2,
    float* __restrict__ scores)
{
  const int lane = threadIdx.x & 31;
  const int m = lane & 15, hi = lane >> 4;

  __shared__ float ls_f[16][CC + 1];   // feat = [t_ratio, enc(12)]

  // pack W1^T as f16 B-tiles (32x16): elem j -> K = j + 16*hi, N = 16*nt + m
  v16h Bw[8];
  #pragma unroll
  for (int nt = 0; nt < 8; ++nt) {
    const int n = nt*16 + m;
    #pragma unroll
    for (int j = 0; j < 16; ++j) {
      const int k = j + 16*hi;
      Bw[nt][j] = (_Float16)((k < CC + 1) ? W1[n*(CC+1) + k] : 0.f);
    }
  }
  float b1v[8], W2v[8];
  #pragma unroll
  for (int nt = 0; nt < 8; ++nt) { const int n = nt*16 + m; b1v[nt] = b1[n]; W2v[nt] = W2[n]; }
  const float b2v = b2[0];

  const int ntiles = (BB * LL) / 16;
  for (int tile = blockIdx.x; tile < ntiles; tile += gridDim.x) {
    if (lane < 16) {
      const int p = tile*16 + lane;
      const int b = p / LL, l = p % LL;
      const float4* pe = reinterpret_cast<const float4*>(enc + (size_t)p * CC);
      const float4 e0 = pe[0], e1 = pe[1], e2 = pe[2];
      ls_f[lane][0]  = fminf((float)l / (float)lengths[b], 1.f);
      ls_f[lane][1]  = e0.x; ls_f[lane][2]  = e0.y; ls_f[lane][3]  = e0.z; ls_f[lane][4]  = e0.w;
      ls_f[lane][5]  = e1.x; ls_f[lane][6]  = e1.y; ls_f[lane][7]  = e1.z; ls_f[lane][8]  = e1.w;
      ls_f[lane][9]  = e2.x; ls_f[lane][10] = e2.y; ls_f[lane][11] = e2.z; ls_f[lane][12] = e2.w;
    }
    __syncthreads();

    // build f16 A-tile 16x32: elem j -> K=j+16*hi
    v16h A;
    #pragma unroll
    for (int j = 0; j < 16; ++j) {
      const int k = j + 16*hi;
      A[j] = (_Float16)((k < CC + 1) ? ls_f[m][k] : 0.f);
    }

    float s[8];
    #pragma unroll
    for (int r = 0; r < 8; ++r) s[r] = 0.f;
    #pragma unroll
    for (int nt = 0; nt < 8; ++nt) {
      v8f c;
      #pragma unroll
      for (int r = 0; r < 8; ++r) c[r] = b1v[nt];
      c = __builtin_amdgcn_wmma_f32_16x16x32_f16(false, A, false, Bw[nt], (short)0, c, false, false);
      #pragma unroll
      for (int r = 0; r < 8; ++r) s[r] += W2v[nt] * ftanh_(c[r]);
    }
    // reduce over the 16 N-lanes of each half
    #pragma unroll
    for (int off = 8; off; off >>= 1) {
      #pragma unroll
      for (int r = 0; r < 8; ++r) s[r] += __shfl_xor(s[r], off, 32);
    }
    if (m == 0) {
      #pragma unroll
      for (int r = 0; r < 8; ++r)
        scores[(size_t)tile*16 + r + 8*hi] = s[r] + b2v;
    }
    __syncthreads();
  }
}

// ---------------------------------------------------------------------------
// Kernel 3: per-row softmax over L and weighted sum  x_enc = attn . enc
// ---------------------------------------------------------------------------
__global__ __launch_bounds__(256) void k_attn_reduce(
    const float* __restrict__ scores, const float* __restrict__ enc,
    float* __restrict__ x_enc)
{
  const int b = blockIdx.x, tid = threadIdx.x;
  __shared__ float red[256];
  __shared__ float outv[CC + 1];

  float mx = -1e30f;
  for (int l = tid; l < LL; l += 256) mx = fmaxf(mx, scores[(size_t)b*LL + l]);
  red[tid] = mx; __syncthreads();
  for (int s = 128; s; s >>= 1) { if (tid < s) red[tid] = fmaxf(red[tid], red[tid+s]); __syncthreads(); }
  mx = red[0]; __syncthreads();

  float se = 0.f, ac[CC];
  #pragma unroll
  for (int c = 0; c < CC; ++c) ac[c] = 0.f;
  for (int l = tid; l < LL; l += 256) {
    const float e = fexp_(scores[(size_t)b*LL + l] - mx);
    se += e;
    const float* pe = enc + ((size_t)b*LL + l)*CC;
    #pragma unroll
    for (int c = 0; c < CC; ++c) ac[c] += e * pe[c];
  }
  for (int q = 0; q < CC + 1; ++q) {
    red[tid] = (q == 0) ? se : ac[q-1];
    __syncthreads();
    for (int s = 128; s; s >>= 1) { if (tid < s) red[tid] += red[tid+s]; __syncthreads(); }
    if (tid == 0) outv[q] = red[0];
    __syncthreads();
  }
  if (tid < CC) x_enc[b*CC + tid] = outv[tid+1] / outv[0];
}

// ---------------------------------------------------------------------------
// Kernel 4: mode LSTM + templates + init-MLP + ird  (tiny tail, one block)
// ---------------------------------------------------------------------------
__global__ __launch_bounds__(64) void k_modes(
    const float* __restrict__ cWih, const float* __restrict__ cWhh,
    const float* __restrict__ cbih, const float* __restrict__ cbhh,
    const float* __restrict__ iW1,  const float* __restrict__ ib1,
    const float* __restrict__ iW2,  const float* __restrict__ ib2,
    const float* __restrict__ mqW,
    float* __restrict__ memb_out, float* __restrict__ ird_out)
{
  const int tid = threadIdx.x;
  __shared__ float sh_h[CC], sh_c[CC], sh_o[CC], sh_g[NG];
  __shared__ float sh_memb[NMODE][CC];
  __shared__ float sh_mre [NMODE][CC][CC];
  __shared__ float sh_marg[NMODE][CC][CC];
  __shared__ float sh_in[2*CC];
  __shared__ float red[64];
  __shared__ float sh_irl[NMODE][CC];
  __shared__ float sh_irest[NMODE];
  __shared__ float sh_core[NMODE][CC];
  __shared__ float sh_restp[NMODE];

  if (tid < CC) { sh_h[tid] = 0.f; sh_c[tid] = 0.f; sh_o[tid] = 0.f; }
  __syncthreads();

  for (int step = 0; step < NMODE; ++step) {
    if (tid < NG) {
      float g = cbih[tid] + cbhh[tid];
      for (int k = 0; k < CC; ++k) g += sh_o[k]*cWih[tid*CC + k] + sh_h[k]*cWhh[tid*CC + k];
      sh_g[tid] = g;
    }
    __syncthreads();
    if (tid < CC) {
      const float ig = fsig_ (sh_g[tid]);
      const float fg = fsig_ (sh_g[tid + CC]);
      const float gg = ftanh_(sh_g[tid + 2*CC]);
      const float og = fsig_ (sh_g[tid + 3*CC]);
      const float c  = fg*sh_c[tid] + ig*gg;
      const float h  = og*ftanh_(c);
      sh_c[tid] = c; sh_h[tid] = h; sh_o[tid] = h;
      sh_memb[step][tid] = h;
    }
    __syncthreads();
  }

  // mre = mode_emb @ mq_W^T  -> (4,12,12)
  for (int idx = tid; idx < NMODE*CC*CC; idx += 64) {
    const int mm = idx / (CC*CC), j = idx % (CC*CC);
    float s = 0.f;
    for (int k = 0; k < CC; ++k) s += sh_memb[mm][k] * mqW[j*CC + k];
    sh_mre[mm][j/CC][j%CC] = s;
  }
  __syncthreads();

  // eql -> softmax over templates -> marg ; ecl computed on the fly from bitmasks
  const unsigned tmask[NT] = {0x091u, 0x089u, 0x049u, 0x491u, 0x891u, 0x489u, 0x249u};
  for (int idx = tid; idx < NMODE*CC; idx += 64) {
    const int mm = idx / CC, r = idx % CC;
    float e[NT]; float emax = -1e30f;
    for (int q = 0; q < NT; ++q) {
      float s = 0.f;
      for (int c = 0; c < CC; ++c) {
        const float qv = ((tmask[q] >> ((c - r + CC) % CC)) & 1u) ? 1.f : -1.f;
        s += sh_mre[mm][r][c] * qv;
      }
      e[q] = s; emax = fmaxf(emax, s);
    }
    float Z = 0.f;
    for (int q = 0; q < NT; ++q) { e[q] = fexp_(e[q] - emax); Z += e[q]; }
    const float iZ = frcp_(Z);
    for (int c = 0; c < CC; ++c) {
      float s = 0.f;
      for (int q = 0; q < NT; ++q) {
        const float qv = ((tmask[q] >> ((c - r + CC) % CC)) & 1u) ? 1.f : -1.f;
        s += e[q] * qv;
      }
      sh_marg[mm][r][c] = s * iZ;
    }
  }
  __syncthreads();

  // init-MLP: 48 core rows + 4 rest rows, 128 hidden each (2 per thread)
  const float b2v = ib2[0];
  for (int rowi = 0; rowi < NMODE*CC + NMODE; ++rowi) {
    const int mm = (rowi < NMODE*CC) ? (rowi / CC) : (rowi - NMODE*CC);
    if (tid < CC) sh_in[tid] = sh_memb[mm][tid];
    else if (tid < 2*CC)
      sh_in[tid] = (rowi < NMODE*CC) ? sh_marg[mm][rowi % CC][tid - CC] : -1.f;
    __syncthreads();
    float p = 0.f;
    #pragma unroll
    for (int hh = 0; hh < 2; ++hh) {
      const int hdx = tid*2 + hh;
      float s = ib1[hdx];
      for (int k = 0; k < 2*CC; ++k) s += sh_in[k] * iW1[hdx*2*CC + k];
      p += iW2[hdx] * ftanh_(s);
    }
    red[tid] = p; __syncthreads();
    for (int s = 32; s; s >>= 1) { if (tid < s) red[tid] += red[tid+s]; __syncthreads(); }
    if (tid == 0) {
      const float v = red[0] + b2v;
      if (rowi < NMODE*CC) sh_irl[mm][rowi % CC] = v; else sh_irest[mm] = v;
    }
    __syncthreads();
  }

  // per-mode softmax over 13 logits -> core/rest
  if (tid < NMODE) {
    float mx = sh_irest[tid];
    for (int c = 0; c < CC; ++c) mx = fmaxf(mx, sh_irl[tid][c]);
    float er[CC];
    float Z = fexp_(sh_irest[tid] - mx);
    for (int c = 0; c < CC; ++c) { er[c] = fexp_(sh_irl[tid][c] - mx); Z += er[c]; }
    const float iZ = frcp_(Z);
    for (int c = 0; c < CC; ++c) sh_core[tid][c] = er[c] * iZ;
    sh_restp[tid] = fexp_(sh_irest[tid] - mx) * iZ;
  }
  __syncthreads();

  for (int idx = tid; idx < NMODE*CC; idx += 64) memb_out[idx] = sh_memb[idx/CC][idx%CC];
  for (int idx = tid; idx < NMODE*CC*(CC+1); idx += 64) {
    const int k = idx / (CC+1), c = idx % (CC+1);
    const int mm = k / CC, i = k % CC;
    ird_out[idx] = (c < CC) ? sh_core[mm][(c - i + CC) % CC] : sh_restp[mm];
  }
}

// ---------------------------------------------------------------------------
// Kernel 5: mode_dist, shift-MLP, key_dist @ ird  -> out (B,13)
// ---------------------------------------------------------------------------
__global__ __launch_bounds__(128) void k_out(
    const float* __restrict__ x_enc, const float* __restrict__ memb,
    const float* __restrict__ ird,
    const float* __restrict__ sW1, const float* __restrict__ sb1,
    const float* __restrict__ sW2, const float* __restrict__ sb2,
    float* __restrict__ out)
{
  const int b = blockIdx.x, tid = threadIdx.x;
  __shared__ float sh_x[CC], sh_m[NMODE][CC], sh_md[NMODE];
  __shared__ float sh_hid[HH];
  __shared__ float sh_shift[NMODE][CC];

  if (tid < CC) sh_x[tid] = x_enc[b*CC + tid];
  if (tid < NMODE*CC) sh_m[tid/CC][tid%CC] = memb[tid];
  __syncthreads();

  if (tid == 0) {
    float lg[NMODE], mx = -1e30f;
    for (int mm = 0; mm < NMODE; ++mm) {
      float s = 0.f;
      for (int c = 0; c < CC; ++c) s += sh_x[c]*sh_m[mm][c];
      lg[mm] = s; mx = fmaxf(mx, s);
    }
    float Z = 0.f;
    for (int mm = 0; mm < NMODE; ++mm) { lg[mm] = fexp_(lg[mm]-mx); Z += lg[mm]; }
    const float iZ = frcp_(Z);
    for (int mm = 0; mm < NMODE; ++mm) sh_md[mm] = lg[mm]*iZ;
  }

  for (int mm = 0; mm < NMODE; ++mm) {
    float s = sb1[tid];
    for (int k = 0; k < CC; ++k) s += sh_m[mm][k] * sW1[tid*2*CC + k];
    for (int k = 0; k < CC; ++k) s += sh_x[k]     * sW1[tid*2*CC + CC + k];
    __syncthreads();              // prior readers of sh_hid done
    sh_hid[tid] = ftanh_(s);
    __syncthreads();
    if (tid < CC) {
      float t = sb2[tid];
      for (int h = 0; h < HH; ++h) t += sh_hid[h] * sW2[tid*HH + h];
      sh_shift[mm][tid] = t;      // logits for now
    }
  }
  __syncthreads();

  if (tid < NMODE) {
    float mx = -1e30f;
    for (int c = 0; c < CC; ++c) mx = fmaxf(mx, sh_shift[tid][c]);
    float Z = 0.f;
    for (int c = 0; c < CC; ++c) { const float e = fexp_(sh_shift[tid][c]-mx); sh_shift[tid][c] = e; Z += e; }
    const float iZ = frcp_(Z);
    for (int c = 0; c < CC; ++c) sh_shift[tid][c] *= iZ;
  }
  __syncthreads();

  if (tid < CC + 1) {
    float s = 0.f;
    for (int k = 0; k < NMODE*CC; ++k) {
      const int mm = k / CC, c = k % CC;
      s += sh_md[mm] * sh_shift[mm][c] * ird[k*(CC+1) + tid];
    }
    out[b*(CC+1) + tid] = s;
  }
}

// ---------------------------------------------------------------------------
extern "C" void kernel_launch(void* const* d_in, const int* in_sizes, int n_in,
                              void* d_out, int out_size, void* d_ws, size_t ws_size,
                              hipStream_t stream) {
  (void)in_sizes; (void)n_in; (void)out_size; (void)ws_size;
  const float* x       = (const float*)d_in[0];
  const int*   lengths = (const int*)  d_in[1];
  const float* localW  = (const float*)d_in[2];
  const float* localb  = (const float*)d_in[3];
  const float* Wih0    = (const float*)d_in[4];
  const float* Whh0    = (const float*)d_in[5];
  const float* bih0    = (const float*)d_in[6];
  const float* bhh0    = (const float*)d_in[7];
  const float* Wih1    = (const float*)d_in[8];
  const float* Whh1    = (const float*)d_in[9];
  const float* bih1    = (const float*)d_in[10];
  const float* bhh1    = (const float*)d_in[11];
  const float* aW1     = (const float*)d_in[12];
  const float* ab1     = (const float*)d_in[13];
  const float* aW2     = (const float*)d_in[14];
  const float* ab2     = (const float*)d_in[15];
  const float* cWih    = (const float*)d_in[16];
  const float* cWhh    = (const float*)d_in[17];
  const float* cbih    = (const float*)d_in[18];
  const float* cbhh    = (const float*)d_in[19];
  const float* sW1     = (const float*)d_in[20];
  const float* sb1     = (const float*)d_in[21];
  const float* sW2     = (const float*)d_in[22];
  const float* sb2     = (const float*)d_in[23];
  const float* iW1     = (const float*)d_in[24];
  const float* ib1     = (const float*)d_in[25];
  const float* iW2     = (const float*)d_in[26];
  const float* ib2     = (const float*)d_in[27];
  const float* mqW     = (const float*)d_in[28];

  float* ws     = (float*)d_ws;
  float* enc    = ws;                          // B*L*C
  float* scores = enc    + (size_t)BB*LL*CC;   // B*L
  float* x_enc  = scores + (size_t)BB*LL;      // B*C
  float* memb   = x_enc  + (size_t)BB*CC;      // 4*12
  float* ird    = memb   + NMODE*CC;           // 48*13

  k_lstm_scan<<<BB/16, 32, 0, stream>>>(x, lengths, localW, localb,
      Wih0, Whh0, bih0, bhh0, Wih1, Whh1, bih1, bhh1, enc);
  k_attn_score<<<8192, 32, 0, stream>>>(enc, lengths, aW1, ab1, aW2, ab2, scores);
  k_attn_reduce<<<BB, 256, 0, stream>>>(scores, enc, x_enc);
  k_modes<<<1, 64, 0, stream>>>(cWih, cWhh, cbih, cbhh, iW1, ib1, iW2, ib2, mqW, memb, ird);
  k_out<<<BB, 128, 0, stream>>>(x_enc, memb, ird, sW1, sb1, sW2, sb2, (float*)d_out);
}
